// Memcodes_41077067219408
// MI455X (gfx1250) — compile-verified, matrix operand order checked by reading, not run on hardware
//
#include <hip/hip_runtime.h>
#include <hip/hip_bf16.h>
#include <math.h>

typedef __attribute__((ext_vector_type(16))) _Float16 v16h;
typedef __attribute__((ext_vector_type(8)))  float    v8f;

#define HEADS 8
#define DIM 768
#define DH 96          // DIM / HEADS
#define CODEBOOK 1024
#define NPOS 1024      // 32*32
#define NPAIR 64       // b * heads
#define B 8
#define KPAD 104       // padded LDS row stride (halfs): 208B -> perfect bank spread

// ---------------- workspace layout (bytes) ----------------
#define WS_K16 0                 // HEADS*CODEBOOK*DH halfs   = 1,572,864 B
#define WS_Q16 1572864           // B*HEADS*NPOS*DH halfs     = 12,582,912 B
#define WS_V32 14155776          // HEADS*CODEBOOK*DH floats  = 3,145,728 B
#define WS_IDX 17301504          // B*HEADS*NPOS ints         = 262,144 B

#define OUT_ELEMS (B * DIM * NPOS)       // 6,291,456
#define IDX_ELEMS (B * HEADS * NPOS)     // 65,536

// Kernel 1: k = codes @ Wk (f16), v = codes @ Wv (f32), per head.
__global__ __launch_bounds__(256) void kv_project(
    const float* __restrict__ codes, const float* __restrict__ Wk,
    const float* __restrict__ Wv, _Float16* __restrict__ k16,
    float* __restrict__ v32) {
  int t = blockIdx.x * 256 + threadIdx.x;        // HEADS*CODEBOOK*DH threads
  int h  = t / (CODEBOOK * DH);
  int r  = t % (CODEBOOK * DH);
  int j  = r / DH;
  int dd = r % DH;
  const float* crow = codes + (h * CODEBOOK + j) * DH;
  const float* wkc  = Wk + h * DH * DH + dd;
  const float* wvc  = Wv + h * DH * DH + dd;
  float ka = 0.f, va = 0.f;
#pragma unroll 8
  for (int e = 0; e < DH; ++e) {
    float ce = crow[e];
    ka = fmaf(ce, wkc[e * DH], ka);
    va = fmaf(ce, wvc[e * DH], va);
  }
  k16[t] = (_Float16)ka;
  v32[t] = va;
}

// Kernel 2: q16[pair][i][d] = x[b][h*96+d][i] * SCALE  (f16)
__global__ __launch_bounds__(256) void q_convert(
    const float* __restrict__ x, _Float16* __restrict__ q16) {
  int t = blockIdx.x * 256 + threadIdx.x;        // OUT_ELEMS threads
  int pair = t / (NPOS * DH);
  int r    = t % (NPOS * DH);
  int i    = r / DH;
  int dd   = r % DH;
  int b = pair / HEADS, h = pair % HEADS;
  const float SCALE = 0.10206207261596575f;      // 96^-0.5
  float qv = x[((size_t)(b * DIM + h * DH + dd)) * NPOS + i] * SCALE;
  q16[t] = (_Float16)qv;
}

// Assemble a v16h from LDS with two 16B-aligned b128 reads.
__device__ __forceinline__ v16h lds_v16h(const _Float16* p) {
  union { uint4 u[2]; v16h v; } t;
  t.u[0] = *(const uint4*)(p);
  t.u[1] = *(const uint4*)(p + 8);
  return t.v;
}

// Kernel 3: per (b,h) pair: logits = q @ k^T via WMMA (k staged in LDS),
// running argmax over the codebook dimension. Each wave owns 32 q-rows
// (two A tiles) so every staged B tile feeds 2 WMMAs -> half the LDS
// read traffic per FLOP.
__global__ __launch_bounds__(256) void logits_argmax(
    const _Float16* __restrict__ q16, const _Float16* __restrict__ k16,
    int* __restrict__ idxws, float* __restrict__ idxout) {
  __shared__ __align__(32) _Float16 kb[128 * KPAD];  // 26,624 B

  int pair   = blockIdx.x >> 2;                  // 64 pairs
  int rowblk = blockIdx.x & 3;                   // 4 row-blocks of 256
  int wave = threadIdx.x >> 5;
  int lane = threadIdx.x & 31;
  int ln = lane & 15;                            // A-row / B-col / C-col select
  int hi = lane >> 4;                            // K half select
  int h = pair & 7;
  int i0 = rowblk * 256 + wave * 32;

  // Two A tile sets: rows i0..i0+15 and i0+16..i0+31, K = 96 in 32-chunks.
  size_t qb0 = ((size_t)pair * NPOS + i0 + ln) * DH + hi * 16;
  size_t qb1 = qb0 + (size_t)16 * DH;
  v16h a0 = *(const v16h*)(q16 + qb0);
  v16h a1 = *(const v16h*)(q16 + qb0 + 32);
  v16h a2 = *(const v16h*)(q16 + qb0 + 64);
  v16h a3 = *(const v16h*)(q16 + qb1);
  v16h a4 = *(const v16h*)(q16 + qb1 + 32);
  v16h a5 = *(const v16h*)(q16 + qb1 + 64);

  float bv0[8], bv1[8];
  int   bi0[8], bi1[8];
#pragma unroll
  for (int r = 0; r < 8; ++r) {
    bv0[r] = -INFINITY; bi0[r] = 0;
    bv1[r] = -INFINITY; bi1[r] = 0;
  }

  const _Float16* khead = k16 + (size_t)h * CODEBOOK * DH;

  for (int jc = 0; jc < 8; ++jc) {               // 8 chunks of 128 codebook rows
    const _Float16* src = khead + (size_t)jc * 128 * DH;
    // Cooperative stage: 128 rows x 96 halfs = 1536 uint4, 6 per thread,
    // fully coalesced reads; padded rows in LDS.
#pragma unroll
    for (int u6 = 0; u6 < 6; ++u6) {
      int u = threadIdx.x + u6 * 256;
      int g = u * 8;                             // half offset in chunk
      int row = g / DH;
      int col = g % DH;
      uint4 d = *(const uint4*)(src + g);
      *(uint4*)(kb + row * KPAD + col) = d;
    }
    __syncthreads();

    // Prefetch next chunk toward L2/WGP$ while we compute this one.
    if (jc < 7) __builtin_prefetch(src + 128 * DH + threadIdx.x * 48, 0, 0);

#pragma unroll
    for (int jt = 0; jt < 8; ++jt) {             // 8 j-tiles of 16 per chunk
      const _Float16* kp = kb + (jt * 16 + ln) * KPAD + hi * 16;
      v16h b0 = lds_v16h(kp);
      v16h b1 = lds_v16h(kp + 32);
      v16h b2 = lds_v16h(kp + 64);
      v8f c0 = {}, c1 = {};
      c0 = __builtin_amdgcn_wmma_f32_16x16x32_f16(false, a0, false, b0,
                                                  (short)0, c0, false, false);
      c1 = __builtin_amdgcn_wmma_f32_16x16x32_f16(false, a3, false, b0,
                                                  (short)0, c1, false, false);
      c0 = __builtin_amdgcn_wmma_f32_16x16x32_f16(false, a1, false, b1,
                                                  (short)0, c0, false, false);
      c1 = __builtin_amdgcn_wmma_f32_16x16x32_f16(false, a4, false, b1,
                                                  (short)0, c1, false, false);
      c0 = __builtin_amdgcn_wmma_f32_16x16x32_f16(false, a2, false, b2,
                                                  (short)0, c0, false, false);
      c1 = __builtin_amdgcn_wmma_f32_16x16x32_f16(false, a5, false, b2,
                                                  (short)0, c1, false, false);
      int col = jc * 128 + jt * 16 + ln;         // this lane's logit column
#pragma unroll
      for (int r = 0; r < 8; ++r) {
        float v0 = c0[r];
        if (v0 > bv0[r]) { bv0[r] = v0; bi0[r] = col; }
        float v1 = c1[r];
        if (v1 > bv1[r]) { bv1[r] = v1; bi1[r] = col; }
      }
    }
    __syncthreads();
  }

  // Reduce across the 16 lanes of each half (xor masks stay within halves).
#pragma unroll
  for (int off = 1; off < 16; off <<= 1) {
#pragma unroll
    for (int r = 0; r < 8; ++r) {
      float ov; int oi;
      ov = __shfl_xor(bv0[r], off, 32);
      oi = __shfl_xor(bi0[r], off, 32);
      if (ov > bv0[r] || (ov == bv0[r] && oi < bi0[r])) { bv0[r] = ov; bi0[r] = oi; }
      ov = __shfl_xor(bv1[r], off, 32);
      oi = __shfl_xor(bi1[r], off, 32);
      if (ov > bv1[r] || (ov == bv1[r] && oi < bi1[r])) { bv1[r] = ov; bi1[r] = oi; }
    }
  }

  // lane 0 -> rows +0..7, lane 16 -> rows +8..15 (per A tile set).
  if (ln == 0) {
#pragma unroll
    for (int r = 0; r < 8; ++r) {
      int i = i0 + hi * 8 + r;
      int o = pair * NPOS + i;
      idxws[o]  = bi0[r];
      idxout[o] = (float)bi0[r];
      int i2 = i + 16;
      int o2 = pair * NPOS + i2;
      idxws[o2]  = bi1[r];
      idxout[o2] = (float)bi1[r];
    }
  }
}

// Kernel 4: out[b][c][i] = v[h][idx[b,h,i]][dd]; coalesced writes.
__global__ __launch_bounds__(256) void gather_out(
    const float* __restrict__ v32, const int* __restrict__ idxws,
    float* __restrict__ out) {
  int t = blockIdx.x * 256 + threadIdx.x;        // OUT_ELEMS threads
  if (t == 0) out[OUT_ELEMS] = 0.0f;             // codebook_loss = 0
  int b = t / (DIM * NPOS);
  int r = t % (DIM * NPOS);
  int cdim = r / NPOS;
  int i    = r % NPOS;
  int h  = cdim / DH;
  int dd = cdim % DH;
  int id = idxws[(b * HEADS + h) * NPOS + i];
  out[t] = v32[((size_t)h * CODEBOOK + id) * DH + dd];
}

extern "C" void kernel_launch(void* const* d_in, const int* in_sizes, int n_in,
                              void* d_out, int out_size, void* d_ws, size_t ws_size,
                              hipStream_t stream) {
  const float* x     = (const float*)d_in[0];
  const float* codes = (const float*)d_in[1];
  const float* Wk    = (const float*)d_in[2];
  const float* Wv    = (const float*)d_in[3];
  float* out = (float*)d_out;

  char* ws = (char*)d_ws;
  _Float16* k16 = (_Float16*)(ws + WS_K16);
  _Float16* q16 = (_Float16*)(ws + WS_Q16);
  float*    v32 = (float*)(ws + WS_V32);
  int*      idxws = (int*)(ws + WS_IDX);
  float* idxout = out + OUT_ELEMS + 1;           // out, loss, indices

  kv_project<<<(HEADS * CODEBOOK * DH) / 256, 256, 0, stream>>>(
      codes, Wk, Wv, k16, v32);
  q_convert<<<OUT_ELEMS / 256, 256, 0, stream>>>(x, q16);
  logits_argmax<<<NPAIR * 4, 256, 0, stream>>>(q16, k16, idxws, idxout);
  gather_out<<<OUT_ELEMS / 256, 256, 0, stream>>>(v32, idxws, out);
}
